// TopoSeg_24455543783750
// MI455X (gfx1250) — compile-verified, matrix operand order checked by reading, not run on hardware
//
#include <hip/hip_runtime.h>
#include <hip/hip_bf16.h>

typedef __attribute__((ext_vector_type(16))) _Float16 v16h;
typedef __attribute__((ext_vector_type(8)))  float    v8f;

#define NUM_THRESH 100
#define WW   64
#define NPIX 4096           // 64*64
#define NTHREADS 256        // 8 wave32 waves
#define PIX_PER_THREAD 16   // 4096 / 256
#define MAX_ITERS 1024      // safety cap; early-exit on convergence

__device__ __forceinline__ int imax2(int a, int b) { return a > b ? a : b; }

__global__ void betti_zero_kernel(float* out, int n) {
    int i = blockIdx.x * blockDim.x + threadIdx.x;
    if (i < n) out[i] = 0.0f;
}

__launch_bounds__(NTHREADS)
__global__ void betti_err_kernel(const float* __restrict__ prob,
                                 const int*   __restrict__ gt,
                                 float*       __restrict__ out)
{
    __shared__ int   lab[NPIX];   // 16 KB label field (0 = background)
    __shared__ int   changed;
    __shared__ float acc[4];      // roots, px, ev, eh

    const int tid  = threadIdx.x;
    const int lane = tid & 31;

    // blockIdx.x encodes (t, b, c): 100 * 8 * 2 = 1600 workgroups
    const int blk = blockIdx.x;
    const int c   = blk & 1;          // channel: 0=inside(+union), 1=boundary
    const int b   = (blk >> 1) & 7;   // image
    const int t   = blk >> 4;         // threshold index
    const float thr = (float)t * (1.0f / 99.0f);   // linspace(0,1,100)

    if (tid < 4) acc[tid] = 0.0f;
    if (tid == 0) changed = 0;

    const float* pr = prob + (size_t)((b * 3 + c) * NPIX);

    // ---- binarize + seed labels (id = p+1 so 0 means background) ----
    unsigned maskbits = 0u;
    #pragma unroll
    for (int k = 0; k < PIX_PER_THREAD; ++k) {
        int p = tid + k * NTHREADS;        // strided: contiguous per sweep
        float v = pr[p];
        int m = (v > thr) ? 1 : 0;
        maskbits |= (unsigned)m << k;
        lab[p] = m ? (p + 1) : 0;
    }
    __syncthreads();

    // ---- in-place monotone max-label propagation to fixed point ----
    for (int it = 0; it < MAX_ITERS; ++it) {
        // invariant: changed == 0 and visible to all threads here
        int any = 0;
        #pragma unroll
        for (int k = 0; k < PIX_PER_THREAD; ++k) {
            if (maskbits & (1u << k)) {
                int p = tid + k * NTHREADS;
                int x = p & (WW - 1);
                int v = lab[p];
                int m = v;
                if (p >= WW)          m = imax2(m, lab[p - WW]);
                if (p <  NPIX - WW)   m = imax2(m, lab[p + WW]);
                if (x >  0)           m = imax2(m, lab[p - 1]);
                if (x <  WW - 1)      m = imax2(m, lab[p + 1]);
                if (m != v) { lab[p] = m; any = 1; }   // monotone: race-safe
            }
        }
        if (any) changed = 1;          // benign multi-writer of same value
        __syncthreads();
        int done = (changed == 0);     // uniform across block
        __syncthreads();
        if (tid == 0) changed = 0;
        if (done) break;               // uniform break
        __syncthreads();               // reset visible before next sweep
    }

    // ---- per-thread partial stats (exact small integers) ----
    float p_roots = 0.0f, p_px = 0.0f, p_ev = 0.0f, p_eh = 0.0f;
    #pragma unroll
    for (int k = 0; k < PIX_PER_THREAD; ++k) {
        if (maskbits & (1u << k)) {
            int p = tid + k * NTHREADS;
            int x = p & (WW - 1);
            p_px += 1.0f;
            if (lab[p] == p + 1) p_roots += 1.0f;                   // component rep
            if (p < NPIX - WW && lab[p + WW] != 0) p_ev += 1.0f;    // vertical edge
            if (x < WW - 1    && lab[p + 1]  != 0) p_eh += 1.0f;    // horizontal edge
        }
    }

    // ---- intra-wave reduction via v_wmma_f32_16x16x32_f16 ----
    // A[m][K]: lane m slots j -> K=j, lane m+16 slots j -> K=16+j.
    // B selector: column q (q<4) is 1 at K==q and K==16+q, so
    // D[:,q] column-sums yield the per-quantity wave totals.
    v16h a, bsel;
    #pragma unroll
    for (int i = 0; i < 16; ++i) { a[i] = (_Float16)0.0f; bsel[i] = (_Float16)0.0f; }
    a[0] = (_Float16)p_roots;
    a[1] = (_Float16)p_px;
    a[2] = (_Float16)p_ev;
    a[3] = (_Float16)p_eh;
    int l15 = lane & 15;
    if (l15 < 4) bsel[l15] = (_Float16)1.0f;

    v8f cacc = {};
    cacc = __builtin_amdgcn_wmma_f32_16x16x32_f16(
        /*neg_a=*/false, a, /*neg_b=*/false, bsel,
        /*c_mod=*/(short)0, cacc, /*reuse_a=*/false, /*reuse_b=*/false);

    // lane n holds D[0..7][n] (n<16) or D[8..15][n-16]; sum the 8 rows held here
    float s = cacc[0] + cacc[1] + cacc[2] + cacc[3]
            + cacc[4] + cacc[5] + cacc[6] + cacc[7];
    // total_q = sum over all 16 rows of column q
    float t0 = __shfl(s, 0, 32) + __shfl(s, 16, 32);
    float t1 = __shfl(s, 1, 32) + __shfl(s, 17, 32);
    float t2 = __shfl(s, 2, 32) + __shfl(s, 18, 32);
    float t3 = __shfl(s, 3, 32) + __shfl(s, 19, 32);

    if (lane == 0) {                       // one wave-leader per 8 waves
        atomicAdd(&acc[0], t0);            // ds_add_f32
        atomicAdd(&acc[1], t1);
        atomicAdd(&acc[2], t2);
        atomicAdd(&acc[3], t3);
    }
    __syncthreads();

    // ---- betti numbers + L1 error vs gt, one global atomic per WG ----
    if (tid == 0) {
        float b0    = acc[0];
        float euler = acc[1] - (acc[2] + acc[3]);
        float b1    = b0 - euler;
        float e;
        if (c == 0) {
            // inside betti compared against gt channel 0 AND channel 2 (union)
            const int* g0 = gt + ((size_t)(b * 3 + 0) * NUM_THRESH + t) * 2;
            const int* g2 = gt + ((size_t)(b * 3 + 2) * NUM_THRESH + t) * 2;
            e = fabsf(b0 - (float)g0[0]) + fabsf(b1 - (float)g0[1])
              + fabsf(b0 - (float)g2[0]) + fabsf(b1 - (float)g2[1]);
        } else {
            const int* g1 = gt + ((size_t)(b * 3 + 1) * NUM_THRESH + t) * 2;
            e = fabsf(b0 - (float)g1[0]) + fabsf(b1 - (float)g1[1]);
        }
        atomicAdd(&out[b], e);             // global_atomic_add_f32
    }
}

extern "C" void kernel_launch(void* const* d_in, const int* in_sizes, int n_in,
                              void* d_out, int out_size, void* d_ws, size_t ws_size,
                              hipStream_t stream)
{
    const float* prob = (const float*)d_in[0];   // [8,3,64,64] fp32
    const int*   gt   = (const int*)d_in[1];     // [8,3,100,2] int32
    float*       out  = (float*)d_out;           // [8] fp32

    betti_zero_kernel<<<1, 64, 0, stream>>>(out, out_size);
    betti_err_kernel<<<NUM_THRESH * 8 * 2, NTHREADS, 0, stream>>>(prob, gt, out);
}
